// RPPatchEmbed_kernel_share_61443802136903
// MI455X (gfx1250) — matrix operand order repaired and checked by reading, not compiled, not run on hardware
//
#include <hip/hip_runtime.h>
#include <hip/hip_bf16.h>

typedef __attribute__((ext_vector_type(16))) _Float16 v16h;
typedef __attribute__((ext_vector_type(8)))  _Float16 v8h;
typedef __attribute__((ext_vector_type(8)))  float    v8f;

union frag16 { v16h v; v8h h[2]; };

#define BB   32
#define EE   768
#define NTOT 588
#define JCUT 392
#define NSEL 196
#define MAXSEL (BB * NSEL)   // 6272

// async copy of 16 bytes global -> LDS (per lane), tracked on ASYNCcnt
__device__ __forceinline__ void async_b128_to_lds(void* lds_dst, const void* gbase,
                                                  unsigned byte_off) {
    unsigned lds_addr = (unsigned)(uintptr_t)lds_dst;  // addr[31:0] == LDS offset
    asm volatile("global_load_async_to_lds_b128 %0, %1, %2"
                 :
                 : "v"(lds_addr), "v"(byte_off), "s"(gbase)
                 : "memory");
}
__device__ __forceinline__ void wait_async0() {
    asm volatile("s_wait_asynccnt 0x0" ::: "memory");
}

// ---------------------------------------------------------------------------
// small utility kernels
// ---------------------------------------------------------------------------
__global__ void k_init(int* cnt) {
    if (threadIdx.x < 3) cnt[threadIdx.x] = 0;
}

__global__ void k_cvt_f16(const float* __restrict__ in, _Float16* __restrict__ out, int n) {
    int i = blockIdx.x * blockDim.x + threadIdx.x;
    if (i < n) out[i] = (_Float16)in[i];
}

// pack (COUT, CIN, 3, 3) f32 weights -> [COUT][KP] f16 (contiguous im2col order)
__global__ void k_pack_w3x3(const float* __restrict__ w, _Float16* __restrict__ out,
                            int cout, int K, int KP) {
    int gid = blockIdx.x * blockDim.x + threadIdx.x;
    if (gid >= cout * KP) return;
    int m = gid / KP, k = gid - m * KP;
    out[gid] = (k < K) ? (_Float16)w[m * K + k] : (_Float16)0.f;
}

// pack w_shared[:, :, :KA, :KA] -> [768][3*KA*KA] f16, k = c*KA*KA + i*KA + j
__global__ void k_pack_ws(const float* __restrict__ w, _Float16* __restrict__ out, int KA) {
    int K = 3 * KA * KA;
    int gid = blockIdx.x * blockDim.x + threadIdx.x;
    if (gid >= EE * K) return;
    int e = gid / K, k = gid - e * K;
    int kk2 = KA * KA;
    int c = k / kk2, rem = k - c * kk2;
    int i = rem / KA, j = rem - i * KA;
    out[gid] = (_Float16)w[((size_t)(e * 3 + c) * 48 + i) * 48 + j];
}

// ---------------------------------------------------------------------------
// implicit-GEMM conv 3x3 stride-2 pad-1 + bias + BN + ReLU, f16 in/out, WMMA
// block = 256 threads = 8 waves; block tile M=64 x N=64
// wave: mt in 0..3 (16 rows), ng in 0..1 (32 cols = 2 wmma tiles, shared A)
// A tile staged with global_load_async_to_lds_b128 (ASYNCcnt);
// B tile im2col-gathered and stored as Bs[n][k] so fragments are ds_load_b128.
// ---------------------------------------------------------------------------
template<int CIN, int COUT, int HIN, int WIN>
__global__ void __launch_bounds__(256)
k_conv3x3s2(const _Float16* __restrict__ in, const _Float16* __restrict__ wpk,
            const float* __restrict__ bias, const float* __restrict__ gam,
            const float* __restrict__ bet, const float* __restrict__ mu,
            const float* __restrict__ var, _Float16* __restrict__ out) {
    constexpr int HOUT = HIN / 2, WOUT = WIN / 2;
    constexpr int NSP  = HOUT * WOUT;
    constexpr int K    = CIN * 9;
    constexpr int KP   = (K + 31) & ~31;

    __shared__ __align__(16) _Float16 As[64][40];
    __shared__ __align__(16) _Float16 Bs[64][40];   // [n][k]

    const int tid  = threadIdx.x;
    const int lane = tid & 31;
    const int wv   = tid >> 5;
    const int mt   = wv & 3;      // M subtile (x16)
    const int ng   = wv >> 2;     // N group   (x32)

    const int nbase = blockIdx.x * 64;
    const int mbase = blockIdx.y * 64;
    const int b     = blockIdx.z;

    const _Float16* inb = in + (size_t)b * CIN * HIN * WIN;

    // per-thread fixed staging slot: 4 threads per column, 8 k each
    const int s_n   = tid >> 2;          // 0..63
    const int s_c8  = (tid & 3) * 8;     // 0,8,16,24
    const int s_pos = nbase + s_n;
    const int s_oy  = s_pos / WOUT, s_ox = s_pos - s_oy * WOUT;
    const bool s_ok = (s_pos < NSP);

    v8f acc0 = {}, acc1 = {};

    for (int k0 = 0; k0 < KP; k0 += 32) {
        // A: 64x32 weights tile, async 16B per lane straight into LDS
        async_b128_to_lds(&As[s_n][s_c8], wpk,
                          (unsigned)(((size_t)(mbase + s_n) * KP + k0 + s_c8) * 2));
        // B: im2col 64 cols x 32 k, 8 k per thread, one b128 LDS store
        {
            v8h tmp;
            #pragma unroll
            for (int u = 0; u < 8; ++u) {
                int k = k0 + s_c8 + u;
                _Float16 val = (_Float16)0.f;
                if (s_ok && k < K) {
                    int c  = k / 9, r9 = k - c * 9;
                    int di = r9 / 3, dj = r9 - di * 3;
                    int iy = s_oy * 2 - 1 + di, ix = s_ox * 2 - 1 + dj;
                    if (iy >= 0 && iy < HIN && ix >= 0 && ix < WIN)
                        val = inb[(size_t)c * HIN * WIN + iy * WIN + ix];
                }
                tmp[u] = val;
            }
            *(v8h*)&Bs[s_n][s_c8] = tmp;
        }
        wait_async0();
        __syncthreads();

        // fragments (ISA 16-bit A 16x32 / B 32x16 lane layouts)
        frag16 a, b0, b1;
        {
            int m  = mt * 16 + (lane & 15);
            int kh = (lane >> 4) * 8;
            a.h[0] = *(const v8h*)&As[m][kh];
            a.h[1] = *(const v8h*)&As[m][16 + kh];
            int n0 = ng * 32 + (lane & 15);
            int kb = (lane >> 4) * 8;
            b0.h[0] = *(const v8h*)&Bs[n0][kb];
            b0.h[1] = *(const v8h*)&Bs[n0][16 + kb];
            b1.h[0] = *(const v8h*)&Bs[n0 + 16][kb];
            b1.h[1] = *(const v8h*)&Bs[n0 + 16][16 + kb];
        }
        acc0 = __builtin_amdgcn_wmma_f32_16x16x32_f16(false, a.v, false, b0.v, (short)0,
                                                      acc0, false, false);
        acc1 = __builtin_amdgcn_wmma_f32_16x16x32_f16(false, a.v, false, b1.v, (short)0,
                                                      acc1, false, false);
        __syncthreads();
    }

    // epilogue: bias + BN + ReLU -> f16 NCHW
    {
        int m0 = mbase + mt * 16 + ((lane >> 4) ? 8 : 0);
        int n0 = nbase + ng * 32 + (lane & 15);
        int n1 = n0 + 16;
        _Float16* ob = out + (size_t)b * COUT * NSP;
        #pragma unroll
        for (int r = 0; r < 8; ++r) {
            int m = m0 + r;
            float al = gam[m] * rsqrtf(var[m] + 1e-5f);
            float bo = (bias[m] - mu[m]) * al + bet[m];
            if (n0 < NSP) {
                float v = acc0[r] * al + bo;
                ob[(size_t)m * NSP + n0] = (_Float16)(v > 0.f ? v : 0.f);
            }
            if (n1 < NSP) {
                float v = acc1[r] * al + bo;
                ob[(size_t)m * NSP + n1] = (_Float16)(v > 0.f ? v : 0.f);
            }
        }
    }
}

// ---------------------------------------------------------------------------
// conv4: 256 -> 3 channels, 3x3 s2 p1, scalar (tiny: 43 MMAC total)
// ---------------------------------------------------------------------------
__global__ void k_score(const _Float16* __restrict__ r3, const float* __restrict__ w4,
                        const float* __restrict__ b4, float* __restrict__ score) {
    int gid = blockIdx.x * blockDim.x + threadIdx.x;
    if (gid >= BB * 3 * 196) return;
    int b   = gid / (3 * 196);
    int rem = gid - b * (3 * 196);
    int a   = rem / 196;
    int pos = rem - a * 196;
    int oy = pos / 14, ox = pos - oy * 14;
    const _Float16* rb = r3 + (size_t)b * 256 * 784;
    float sum = b4[a];
    for (int c = 0; c < 256; ++c) {
        const _Float16* rc = rb + c * 784;
        const float*    wc = w4 + (a * 256 + c) * 9;
        #pragma unroll
        for (int di = 0; di < 3; ++di) {
            int iy = oy * 2 - 1 + di;
            if (iy < 0 || iy >= 28) continue;
            #pragma unroll
            for (int dj = 0; dj < 3; ++dj) {
                int ix = ox * 2 - 1 + dj;
                if (ix < 0 || ix >= 28) continue;
                sum += (float)rc[iy * 28 + ix] * wc[di * 3 + dj];
            }
        }
    }
    score[b * NTOT + a * 196 + pos] = sum;
}

// ---------------------------------------------------------------------------
// per-batch: stable ascending bitonic sort, softmax, STE weight, per-scale
// compaction of the 196 selected patches. One block per batch element.
// ---------------------------------------------------------------------------
__global__ void k_select(const float* __restrict__ score, int* __restrict__ cnt,
                         int* __restrict__ list, float* __restrict__ wl) {
    const int b   = blockIdx.x;
    const int tid = threadIdx.x;
    __shared__ float key[1024];
    __shared__ int   idx[1024];
    __shared__ float partial[256];

    const float* s = score + b * NTOT;
    for (int i = tid; i < 1024; i += blockDim.x) {
        key[i] = (i < NTOT) ? s[i] : 3.0e38f;
        idx[i] = i;
    }
    __syncthreads();

    // bitonic, lexicographic (key, idx) => stable like jnp.argsort
    for (int k = 2; k <= 1024; k <<= 1) {
        for (int j = k >> 1; j > 0; j >>= 1) {
            for (int i = tid; i < 1024; i += blockDim.x) {
                int l = i ^ j;
                if (l > i) {
                    bool asc = ((i & k) == 0);
                    float ki = key[i], kl = key[l];
                    int   ii = idx[i], il = idx[l];
                    bool  gt = (ki > kl) || (ki == kl && ii > il);
                    if (gt == asc) { key[i] = kl; key[l] = ki; idx[i] = il; idx[l] = ii; }
                }
            }
            __syncthreads();
        }
    }

    const float mx = key[NTOT - 1];  // largest real score (pads sort above)
    float ps = 0.f;
    for (int i = tid; i < NTOT; i += blockDim.x) ps += expf(key[i] - mx);
    partial[tid] = ps;
    __syncthreads();
    for (int st = 128; st > 0; st >>= 1) {
        if (tid < st) partial[tid] += partial[tid + st];
        __syncthreads();
    }
    const float inv = 1.f / partial[0];

    for (int j = tid; j < NSEL; j += blockDim.x) {
        int   t  = idx[JCUT + j];
        float sm = expf(key[JCUT + j] - mx) * inv;
        float wt = (1.0f - sm) + sm;        // STE forward value, as reference computes it
        int a = t / 196, pos = t - a * 196;
        int slot = atomicAdd(&cnt[a], 1);   // slot order nondet; output keyed by (b,j)
        list[a * MAXSEL + slot] = (b << 16) | (j << 8) | pos;
        wl[a * MAXSEL + slot]   = wt;
    }
}

// ---------------------------------------------------------------------------
// selected patch-embed GEMM per scale: M=768, N=cnt[scale], K=3*KA*KA, WMMA
// same 64x64 block tile / 8-wave layout as the conv kernel. The B gather is
// vectorized: each 8-k chunk lies in one image row (KA % 8 == 0, x0 % 8 == 0),
// so in-bounds chunks are a single 16B-aligned global_load_b128.
// ---------------------------------------------------------------------------
template<int KA, int PADA>
__global__ void __launch_bounds__(256)
k_embed(const _Float16* __restrict__ xh, const _Float16* __restrict__ wpk,
        const int* __restrict__ cnt, const int* __restrict__ list,
        const float* __restrict__ wl, float* __restrict__ out, int scale) {
    constexpr int K = 3 * KA * KA;

    __shared__ __align__(16) _Float16 As[64][40];
    __shared__ __align__(16) _Float16 Bs[64][40];   // [n][k]
    __shared__ int   c_b[64], c_oj[64], c_y0[64], c_x0[64];
    __shared__ float c_w[64];

    const int np    = cnt[scale];
    const int nbase = blockIdx.x * 64;
    if (nbase >= np) return;           // uniform across block
    const int mbase = blockIdx.y * 64;
    const int tid = threadIdx.x, lane = tid & 31, wv = tid >> 5;
    const int mt = wv & 3, ng = wv >> 2;

    if (tid < 64) {
        int p = nbase + tid;
        if (p < np) {
            int code = list[scale * MAXSEL + p];
            int bb  = (code >> 16) & 0xff;
            int oj  = (code >> 8) & 0xff;
            int pos = code & 0xff;
            int gh = pos / 14, gw = pos - gh * 14;
            c_b[tid]  = bb;
            c_oj[tid] = oj;
            c_y0[tid] = gh * 16 - PADA;
            c_x0[tid] = gw * 16 - PADA;
            c_w[tid]  = wl[scale * MAXSEL + p];
        } else {
            c_b[tid] = -1;
        }
    }
    __syncthreads();

    const int s_n  = tid >> 2;
    const int s_c8 = (tid & 3) * 8;

    v8f acc0 = {}, acc1 = {};
    for (int k0 = 0; k0 < K; k0 += 32) {
        // A: 64x32 weights tile, async 16B per lane straight into LDS
        async_b128_to_lds(&As[s_n][s_c8], wpk,
                          (unsigned)(((size_t)(mbase + s_n) * K + k0 + s_c8) * 2));
        // B: selected-patch pixels; 8 k per thread = one row-chunk of x
        {
            int bb = c_b[s_n];
            int y0 = c_y0[s_n], x0 = c_x0[s_n];
            int k   = k0 + s_c8;
            int c   = k / (KA * KA);
            int rem = k - c * (KA * KA);
            int i   = rem / KA, jj0 = rem - i * KA;   // jj0 multiple of 8
            int iy  = y0 + i, ix0 = x0 + jj0;
            v8h tmp;
            if (bb >= 0 && iy >= 0 && iy < 224 && ix0 >= 0 && ix0 + 8 <= 224) {
                tmp = *(const v8h*)&xh[(((size_t)bb * 3 + c) * 224 + iy) * 224 + ix0];
            } else {
                #pragma unroll
                for (int u = 0; u < 8; ++u) {
                    int ix = ix0 + u;
                    _Float16 val = (_Float16)0.f;
                    if (bb >= 0 && iy >= 0 && iy < 224 && ix >= 0 && ix < 224)
                        val = xh[(((size_t)bb * 3 + c) * 224 + iy) * 224 + ix];
                    tmp[u] = val;
                }
            }
            *(v8h*)&Bs[s_n][s_c8] = tmp;
        }
        wait_async0();
        __syncthreads();

        frag16 a, b0, b1;
        {
            int m  = mt * 16 + (lane & 15);
            int kh = (lane >> 4) * 8;
            a.h[0] = *(const v8h*)&As[m][kh];
            a.h[1] = *(const v8h*)&As[m][16 + kh];
            int n0 = ng * 32 + (lane & 15);
            int kb = (lane >> 4) * 8;
            b0.h[0] = *(const v8h*)&Bs[n0][kb];
            b0.h[1] = *(const v8h*)&Bs[n0][16 + kb];
            b1.h[0] = *(const v8h*)&Bs[n0 + 16][kb];
            b1.h[1] = *(const v8h*)&Bs[n0 + 16][16 + kb];
        }
        acc0 = __builtin_amdgcn_wmma_f32_16x16x32_f16(false, a.v, false, b0.v, (short)0,
                                                      acc0, false, false);
        acc1 = __builtin_amdgcn_wmma_f32_16x16x32_f16(false, a.v, false, b1.v, (short)0,
                                                      acc1, false, false);
        __syncthreads();
    }

    {
        int m0  = mbase + mt * 16 + ((lane >> 4) ? 8 : 0);
        int nn0 = ng * 32 + (lane & 15);
        int nn1 = nn0 + 16;
        int bb0 = c_b[nn0], bb1 = c_b[nn1];
        if (bb0 >= 0) {
            float w = c_w[nn0];
            float* ob = out + ((size_t)bb0 * NSEL + c_oj[nn0]) * EE + m0;
            #pragma unroll
            for (int r = 0; r < 8; ++r) ob[r] = acc0[r] * w;
        }
        if (bb1 >= 0) {
            float w = c_w[nn1];
            float* ob = out + ((size_t)bb1 * NSEL + c_oj[nn1]) * EE + m0;
            #pragma unroll
            for (int r = 0; r < 8; ++r) ob[r] = acc1[r] * w;
        }
    }
}

// ---------------------------------------------------------------------------
// host orchestration
// ---------------------------------------------------------------------------
extern "C" void kernel_launch(void* const* d_in, const int* in_sizes, int n_in,
                              void* d_out, int out_size, void* d_ws, size_t ws_size,
                              hipStream_t stream) {
    const float* x   = (const float*)d_in[0];
    const float* wsh = (const float*)d_in[1];
    const float* w1  = (const float*)d_in[2];
    const float* b1  = (const float*)d_in[3];
    const float* g1  = (const float*)d_in[4];
    const float* be1 = (const float*)d_in[5];
    const float* m1  = (const float*)d_in[6];
    const float* v1  = (const float*)d_in[7];
    const float* w2  = (const float*)d_in[8];
    const float* b2  = (const float*)d_in[9];
    const float* g2  = (const float*)d_in[10];
    const float* be2 = (const float*)d_in[11];
    const float* m2  = (const float*)d_in[12];
    const float* v2  = (const float*)d_in[13];
    const float* w3  = (const float*)d_in[14];
    const float* b3  = (const float*)d_in[15];
    const float* g3  = (const float*)d_in[16];
    const float* be3 = (const float*)d_in[17];
    const float* m3  = (const float*)d_in[18];
    const float* v3  = (const float*)d_in[19];
    const float* w4  = (const float*)d_in[20];
    const float* b4  = (const float*)d_in[21];
    float* out = (float*)d_out;

    char* p = (char*)d_ws;
    auto alloc = [&](size_t bytes) -> void* {
        void* r = (void*)p;
        p += (bytes + 255) & ~(size_t)255;
        return r;
    };
    const int NX = BB * 3 * 224 * 224;
    _Float16* xh   = (_Float16*)alloc((size_t)NX * 2);
    _Float16* w1p  = (_Float16*)alloc((size_t)64 * 32 * 2);
    _Float16* w2p  = (_Float16*)alloc((size_t)128 * 576 * 2);
    _Float16* w3p  = (_Float16*)alloc((size_t)256 * 1152 * 2);
    _Float16* ws0  = (_Float16*)alloc((size_t)EE * 768 * 2);
    _Float16* ws1  = (_Float16*)alloc((size_t)EE * 3072 * 2);
    _Float16* ws2  = (_Float16*)alloc((size_t)EE * 6912 * 2);
    _Float16* r1   = (_Float16*)alloc((size_t)BB * 64 * 112 * 112 * 2);
    _Float16* r2   = (_Float16*)alloc((size_t)BB * 128 * 56 * 56 * 2);
    _Float16* r3   = (_Float16*)alloc((size_t)BB * 256 * 28 * 28 * 2);
    float*    sc   = (float*)alloc((size_t)BB * NTOT * 4);
    int*      cnt  = (int*)alloc(3 * 4);
    int*      list = (int*)alloc((size_t)3 * MAXSEL * 4);
    float*    wlst = (float*)alloc((size_t)3 * MAXSEL * 4);

    k_init<<<1, 32, 0, stream>>>(cnt);
    k_cvt_f16<<<(NX + 255) / 256, 256, 0, stream>>>(x, xh, NX);

    k_pack_w3x3<<<(64 * 32 + 255) / 256, 256, 0, stream>>>(w1, w1p, 64, 27, 32);
    k_pack_w3x3<<<(128 * 576 + 255) / 256, 256, 0, stream>>>(w2, w2p, 128, 576, 576);
    k_pack_w3x3<<<(256 * 1152 + 255) / 256, 256, 0, stream>>>(w3, w3p, 256, 1152, 1152);
    k_pack_ws<<<(EE * 768 + 255) / 256, 256, 0, stream>>>(wsh, ws0, 16);
    k_pack_ws<<<(EE * 3072 + 255) / 256, 256, 0, stream>>>(wsh, ws1, 32);
    k_pack_ws<<<(EE * 6912 + 255) / 256, 256, 0, stream>>>(wsh, ws2, 48);

    // scorer stack (implicit-GEMM WMMA), 64x64 output tile per block
    k_conv3x3s2<3, 64, 224, 224><<<dim3(196, 1, BB), 256, 0, stream>>>(
        xh, w1p, b1, g1, be1, m1, v1, r1);
    k_conv3x3s2<64, 128, 112, 112><<<dim3(49, 2, BB), 256, 0, stream>>>(
        r1, w2p, b2, g2, be2, m2, v2, r2);
    k_conv3x3s2<128, 256, 56, 56><<<dim3(13, 4, BB), 256, 0, stream>>>(
        r2, w3p, b3, g3, be3, m3, v3, r3);

    k_score<<<(BB * 3 * 196 + 255) / 256, 256, 0, stream>>>(r3, w4, b4, sc);
    k_select<<<BB, 256, 0, stream>>>(sc, cnt, list, wlst);

    // only the selected patches get embedded (<= 6272 across 3 scales)
    dim3 eg(MAXSEL / 64, EE / 64, 1);  // (98, 12)
    k_embed<16, 0><<<eg, 256, 0, stream>>>(xh, ws0, cnt, list, wlst, out, 0);
    k_embed<32, 8><<<eg, 256, 0, stream>>>(xh, ws1, cnt, list, wlst, out, 1);
    k_embed<48, 16><<<eg, 256, 0, stream>>>(xh, ws2, cnt, list, wlst, out, 2);

    (void)in_sizes; (void)n_in; (void)out_size; (void)ws_size;
}
